// BinarySparseCellNet_91010357002757
// MI455X (gfx1250) — matrix operand order repaired and checked by reading, not compiled, not run on hardware
//
#include <hip/hip_runtime.h>
#include <hip/hip_bf16.h>

// ---------------------------------------------------------------------------
// BinarySparseCellNet on MI455X (gfx1250).
//  * all 3x3x3 / 1x1 / strided-down convs: implicit-GEMM via
//    v_wmma_f32_16x16x32_bf16 (bf16 in, fp32 accum), one 16x16 tile per wave32
//  * conv kernel fully templated on shape -> all tile math folds to constants
//  * weights repacked once per launch straight into the WMMA B-fragment
//    layout [coT][tap][chunk][lane][e] -> one contiguous 32B load per lane
//  * activations feeding convs stored as bf16 (converted once in the fused
//    BN+ReLU+mask pass); A-fragments load as 2x16B vector loads
//  * conv outputs / residuals / BN stats stay fp32
//  * conv-transpose (stride==kernel, non-overlapping) as plain VALU dot
// ---------------------------------------------------------------------------

typedef __attribute__((ext_vector_type(16))) __bf16 v16bf;
typedef __attribute__((ext_vector_type(8)))  float  v8f;
typedef __attribute__((ext_vector_type(8)))  unsigned short us8;

#define BN_EPS 1e-4f

__device__ __forceinline__ unsigned short f2bf(float x) {
  unsigned int u = __float_as_uint(x);
  u += 0x7FFFu + ((u >> 16) & 1u);          // round-to-nearest-even
  return (unsigned short)(u >> 16);
}
__device__ __forceinline__ float bf2f(unsigned short u) {
  return __uint_as_float(((unsigned int)u) << 16);
}

union FragBF { v16bf v; unsigned short u[16]; us8 w8[2]; };

// ---------------------------- utility kernels ------------------------------

__global__ void zero_f(float* p, long n) {
  long i = (long)blockIdx.x * blockDim.x + threadIdx.x;
  long st = (long)gridDim.x * blockDim.x;
  for (; i < n; i += st) p[i] = 0.f;
}
__global__ void zero_us(unsigned short* p, long n) {
  long i = (long)blockIdx.x * blockDim.x + threadIdx.x;
  long st = (long)gridDim.x * blockDim.x;
  for (; i < n; i += st) p[i] = 0;
}
__global__ void cvt_k(const float* __restrict__ x, unsigned short* __restrict__ y, long n) {
  long i = (long)blockIdx.x * blockDim.x + threadIdx.x;
  long st = (long)gridDim.x * blockDim.x;
  for (; i < n; i += st) y[i] = f2bf(x[i]);
}

__global__ void scatter_k(const float* __restrict__ f, const int* __restrict__ c,
                          unsigned short* __restrict__ dense, float* __restrict__ m0, int N) {
  int i = blockIdx.x * blockDim.x + threadIdx.x;
  if (i >= N) return;
  int d = c[4 * i], h = c[4 * i + 1], w = c[4 * i + 2];
  long vox = ((long)d * 64 + h) * 64 + w;
  dense[vox * 3 + 0] = f2bf(f[3 * i + 0]);
  dense[vox * 3 + 1] = f2bf(f[3 * i + 1]);
  dense[vox * 3 + 2] = f2bf(f[3 * i + 2]);
  m0[vox] = 1.f;
}

// float OIDHW -> bf16 in WMMA B-fragment order:
// dst[(((coT*K + tap)*NCH + cch)*32 + lane)*16 + e]
//   where co = coT*16 + (lane&15), ci = cch*32 + (e>>3)*16 + (lane>>4)*8 + (e&7)
// (pad region for Ci not multiple of 32 must be pre-zeroed)
__global__ void repack_k(const float* __restrict__ w, unsigned short* __restrict__ o,
                         int Co, int Ci, int K, int NCH) {
  long n = (long)Co * Ci * K;
  long i = (long)blockIdx.x * blockDim.x + threadIdx.x;
  long st = (long)gridDim.x * blockDim.x;
  for (; i < n; i += st) {
    int co = (int)(i / ((long)Ci * K));
    int r  = (int)(i % ((long)Ci * K));
    int ci = r / K;
    int tap = r % K;
    int coT = co >> 4, nn = co & 15;
    int cch = ci >> 5, rr = ci & 31;
    int hi = (rr >> 3) & 1, e = (rr & 7) + ((rr >> 4) << 3);
    int lane = hi * 16 + nn;
    o[((((long)coT * K + tap) * NCH + cch) * 32 + lane) * 16 + e] = f2bf(w[i]);
  }
}

__global__ void mask_sum_k(const float* __restrict__ m, float* dst, long n) {
  __shared__ float s;
  if (threadIdx.x == 0) s = 0.f;
  __syncthreads();
  float a = 0.f;
  long i = (long)blockIdx.x * blockDim.x + threadIdx.x;
  long st = (long)gridDim.x * blockDim.x;
  for (; i < n; i += st) a += m[i];
  atomicAdd(&s, a);
  __syncthreads();
  if (threadIdx.x == 0) atomicAdd(dst, s);
}

// per-channel sum / sumsq (inputs are pre-masked -> plain sums are masked sums)
__global__ void bn_stats_k(const float* __restrict__ x, float* __restrict__ sums,
                           long n, int C) {
  __shared__ float s1[64], s2[64];
  for (int i = threadIdx.x; i < C; i += blockDim.x) { s1[i] = 0.f; s2[i] = 0.f; }
  __syncthreads();
  long i = (long)blockIdx.x * blockDim.x + threadIdx.x;
  long st = (long)gridDim.x * blockDim.x;
  for (; i < n; i += st) {
    float v = x[i];
    int c = (int)(i % C);
    atomicAdd(&s1[c], v);
    atomicAdd(&s2[c], v * v);
  }
  __syncthreads();
  for (int i = threadIdx.x; i < C; i += blockDim.x) {
    atomicAdd(&sums[i], s1[i]);
    atomicAdd(&sums[64 + i], s2[i]);
  }
}

// y(bf16) = relu((x - mu) * g * rsqrt(var + eps) + b) * mask
__global__ void bn_apply_k(const float* __restrict__ x, unsigned short* __restrict__ y,
                           const float* __restrict__ sums,
                           const float* __restrict__ g, const float* __restrict__ b,
                           const float* __restrict__ mask, const float* __restrict__ nPtr,
                           long V, int C) {
  long n = V * C;
  float nn = nPtr[0];
  long i = (long)blockIdx.x * blockDim.x + threadIdx.x;
  long st = (long)gridDim.x * blockDim.x;
  for (; i < n; i += st) {
    int c = (int)(i % C);
    long vox = i / C;
    float mu = sums[c] / nn;
    float var = sums[64 + c] / nn - mu * mu;
    float sc = g[c] * rsqrtf(var + BN_EPS);
    float v = (x[i] - mu) * sc + b[c];
    v = v > 0.f ? v : 0.f;
    y[i] = f2bf(v * mask[vox]);
  }
}

// m2 = reduce_window(m, add, (4,2,2)) > 0
__global__ void down_mask_k(const float* __restrict__ m, float* __restrict__ m2,
                            int Hi, int Wi, int Do, int Ho, int Wo) {
  long n = (long)Do * Ho * Wo;
  long i = (long)blockIdx.x * blockDim.x + threadIdx.x;
  long st = (long)gridDim.x * blockDim.x;
  for (; i < n; i += st) {
    int w = (int)(i % Wo);
    long t = i / Wo;
    int h = (int)(t % Ho);
    int d = (int)(t / Ho);
    float s = 0.f;
    for (int a = 0; a < 4; ++a)
      for (int b = 0; b < 2; ++b)
        for (int c = 0; c < 2; ++c)
          s += m[(((long)(d * 4 + a) * Hi + (h * 2 + b)) * Wi + (w * 2 + c))];
    m2[i] = (s > 0.f) ? 1.f : 0.f;
  }
}

__global__ void copy_ch_k(const float* __restrict__ src, float* __restrict__ dst,
                          long V, int Cs, int dstStride, int off) {
  long n = V * Cs;
  long i = (long)blockIdx.x * blockDim.x + threadIdx.x;
  long st = (long)gridDim.x * blockDim.x;
  for (; i < n; i += st) {
    long vox = i / Cs;
    int c = (int)(i % Cs);
    dst[vox * dstStride + off + c] = src[i];
  }
}

// conv_transpose, stride == kernel == (4,2,2), VALID => non-overlapping scatter
// kernel taps spatially flipped (lax.conv_transpose, transpose_kernel=False)
__global__ void up_conv_k(const unsigned short* __restrict__ in, const float* __restrict__ w,
                          const float* __restrict__ mask, float* __restrict__ out,
                          int Hi, int Wi, int Ci, int Do, int Ho, int Wo, int Co,
                          int outStride, int coOff) {
  long n = (long)Do * Ho * Wo * Co;
  long i = (long)blockIdx.x * blockDim.x + threadIdx.x;
  long st = (long)gridDim.x * blockDim.x;
  for (; i < n; i += st) {
    int co = (int)(i % Co);
    long vox = i / Co;
    int w_ = (int)(vox % Wo);
    long t = vox / Wo;
    int h_ = (int)(t % Ho);
    int d_ = (int)(t / Ho);
    int id = d_ >> 2, ih = h_ >> 1, iw = w_ >> 1;
    int kd = 3 - (d_ & 3), kh = 1 - (h_ & 1), kw = 1 - (w_ & 1);
    const unsigned short* ip = in + (((long)id * Hi + ih) * Wi + iw) * Ci;
    const float* wp = w + (long)co * Ci * 16 + kd * 4 + kh * 2 + kw;
    float s = 0.f;
    for (int ci = 0; ci < Ci; ++ci) s += bf2f(ip[ci]) * wp[(long)ci * 16];
    out[vox * outStride + coOff + co] = s * mask[vox];
  }
}

// gather at coords, 16-ch dot, sigmoid
__global__ void head_k(const unsigned short* __restrict__ x, const int* __restrict__ c,
                       const float* __restrict__ lw, const float* __restrict__ lb,
                       float* __restrict__ out, int N) {
  int i = blockIdx.x * blockDim.x + threadIdx.x;
  if (i >= N) return;
  int d = c[4 * i], h = c[4 * i + 1], w = c[4 * i + 2];
  long vox = ((long)d * 64 + h) * 64 + w;
  float s = lb[0];
  for (int ch = 0; ch < 16; ++ch) s += bf2f(x[vox * 16 + ch]) * lw[ch];
  out[i] = 1.f / (1.f + expf(-s));
}

// ------------------------- implicit-GEMM WMMA conv -------------------------
// in : bf16, voxel-major channels-last [D*H*W, Ci]
// wgt: bf16, pre-swizzled B-fragment order [coT][tap][chunk][lane][16]
// out[vox*outStride + coOff + co] = (conv) * mask[vox] (+ addSrc[vox*Co+co])
// One wave (32-thread block) per 16(voxel)x16(co) tile.
// grid = ((Wo/16)*(Co/16), Ho, Do); all shape math compile-time.
template <int Di, int Hi, int Wi, int Ci, int Do, int Ho, int Wo, int Co,
          int KD, int KH, int KW, int SD, int SH, int SW, int PD, int PH, int PW>
__global__ __launch_bounds__(32) void conv_wmma_t(
    const unsigned short* __restrict__ in, const unsigned short* __restrict__ wgt,
    const float* __restrict__ mask, const float* __restrict__ addSrc,
    float* __restrict__ out, int outStride, int coOff) {
  constexpr int WT   = Wo >> 4;
  constexpr int NTAP = KD * KH * KW;
  constexpr int NCH  = (Ci + 31) >> 5;
  int lane = threadIdx.x;
  int coT = (int)blockIdx.x / WT;
  int wt  = (int)blockIdx.x - coT * WT;
  int h0  = blockIdx.y;
  int d0  = blockIdx.z;
  int woB = wt << 4;
  int m   = lane & 15;                     // A-row (voxel) / D-column index
  int hiS = lane >> 4;
  int co  = (coT << 4) + m;                // D column

  v8f acc = {0.f, 0.f, 0.f, 0.f, 0.f, 0.f, 0.f, 0.f};
  const unsigned short* wbase =
      wgt + ((long)coT * NTAP * NCH * 32 + lane) * 16;

  for (int tap = 0; tap < NTAP; ++tap) {
    int kd = tap / (KH * KW);
    int rm = tap - kd * (KH * KW);
    int kh = rm / KW;
    int kw = rm - kh * KW;
    int di = d0 * SD - PD + kd;
    int hy = h0 * SH - PH + kh;
    int wx = (woB + m) * SW - PW + kw;
    bool ok = (di >= 0) & (di < Di) & (hy >= 0) & (hy < Hi) & (wx >= 0) & (wx < Wi);
    const unsigned short* ap = in + (((long)di * Hi + hy) * Wi + wx) * Ci;
    const unsigned short* wp = wbase + (long)tap * NCH * 512;
#pragma unroll
    for (int cch = 0; cch < NCH; ++cch) {
      const int k0 = cch * 32;
      FragBF a, b;
      // A 16x32 bf16: lane(0-15) VGPR0-3:K0..7, VGPR4-7:K16..23; lanes 16-31 +8
      if constexpr ((Ci & 15) == 0) {
        // runs are 16B-aligned: Ci in {16,32,48,64} -> 2 vector loads
        const us8 z = {0, 0, 0, 0, 0, 0, 0, 0};
        a.w8[0] = (ok && (k0 < Ci)) ? *(const us8*)(ap + k0 + hiS * 8) : z;
        a.w8[1] = (ok && (k0 + 16 < Ci)) ? *(const us8*)(ap + k0 + 16 + hiS * 8) : z;
      } else {
#pragma unroll
        for (int e = 0; e < 16; ++e) {
          const int v  = e >> 1;
          const int kb = (v < 4) ? (v << 1) : (16 + ((v - 4) << 1));
          const int ci = k0 + kb + (e & 1) + hiS * 8;
          a.u[e] = (ok && ci < Ci) ? ap[ci] : (unsigned short)0;
        }
      }
      // B: pre-swizzled fragment -> one contiguous 32B run per lane
      b.v = *(const v16bf*)(wp + (long)cch * 512);
      acc = __builtin_amdgcn_wmma_f32_16x16x32_bf16(
          false, a.v, false, b.v, (short)0, acc, false, false);
    }
  }

  long rowBase = ((long)d0 * Ho + h0) * Wo + woB;
#pragma unroll
  for (int r = 0; r < 8; ++r) {
    int mm = r + hiS * 8;                  // D row: VGPR r -> M = r + hi*8
    long vox = rowBase + mm;
    float v = acc[r];
    if (mask)   v *= mask[vox];
    if (addSrc) v += addSrc[vox * Co + co];
    out[vox * outStride + coOff + co] = v;
  }
}

// ------------------------------ host helpers -------------------------------

static inline int gsz(long n) { return (int)((n + 255) / 256); }

static void stats(const float* x, float* SUMS, long V, int C, hipStream_t s) {
  zero_f<<<1, 128, 0, s>>>(SUMS, 128);
  long n = V * C;
  int g = gsz(n); if (g > 4096) g = 4096;
  bn_stats_k<<<g, 256, 0, s>>>(x, SUMS, n, C);
}
static void bnApply(const float* x, unsigned short* y, const float* SUMS,
                    const float* g, const float* b, const float* mask,
                    const float* nP, long V, int C, hipStream_t s) {
  bn_apply_k<<<gsz(V * C), 256, 0, s>>>(x, y, SUMS, g, b, mask, nP, V, C);
}

// one residual block: OUT = sc + conv2(bnrelu(conv1(bnrelu(X))))
template <int D, int H, int W, int Cin, int Cout>
static void run_block(const float* X, unsigned short* Ta, float* U, unsigned short* Tb,
                      float* SC, const unsigned short* wNin, const unsigned short* w1,
                      const unsigned short* w2,
                      const float* g1, const float* b1, const float* g2, const float* b2,
                      const float* mask, const float* nP, float* SUMS, float* OUT,
                      hipStream_t stream) {
  constexpr long V = (long)D * H * W;
  const dim3 grid((W / 16) * (Cout / 16), H, D);
  const float* addSrc = X;
  if (wNin) {  // 1x1 "nin" shortcut conv on raw X (Ta free until bnApply below)
    cvt_k<<<gsz(V * Cin), 256, 0, stream>>>(X, Ta, V * Cin);
    conv_wmma_t<D, H, W, Cin, D, H, W, Cout, 1, 1, 1, 1, 1, 1, 0, 0, 0>
        <<<grid, 32, 0, stream>>>(Ta, wNin, nullptr, nullptr, SC, Cout, 0);
    addSrc = SC;
  }
  stats(X, SUMS, V, Cin, stream);
  bnApply(X, Ta, SUMS, g1, b1, mask, nP, V, Cin, stream);
  conv_wmma_t<D, H, W, Cin, D, H, W, Cout, 3, 3, 3, 1, 1, 1, 1, 1, 1>
      <<<grid, 32, 0, stream>>>(Ta, w1, mask, nullptr, U, Cout, 0);
  stats(U, SUMS, V, Cout, stream);
  bnApply(U, Tb, SUMS, g2, b2, mask, nP, V, Cout, stream);
  conv_wmma_t<D, H, W, Cout, D, H, W, Cout, 3, 3, 3, 1, 1, 1, 1, 1, 1>
      <<<grid, 32, 0, stream>>>(Tb, w2, mask, addSrc, OUT, Cout, 0);
}

// ------------------------------ orchestration ------------------------------

extern "C" void kernel_launch(void* const* d_in, const int* in_sizes, int n_in,
                              void* d_out, int out_size, void* d_ws, size_t ws_size,
                              hipStream_t stream) {
  (void)n_in; (void)ws_size; (void)out_size;
  // Input flattening assumption: depth-first, dict insertion order, lists in order.
  const float* F = (const float*)d_in[79];
  const int*   C = (const int*)d_in[80];
  int N = in_sizes[80] / 4;

  const long V0 = 64L * 64 * 64;
  const long V1 = 16L * 32 * 32;
  const long V2 = 4L * 16 * 16;

  // --- workspace bump allocator ---
  size_t off = 0;
  auto allocF = [&](long nf) -> float* {
    float* p = (float*)((char*)d_ws + off);
    off += (((size_t)nf * 4 + 255) / 256) * 256;
    return p;
  };
  auto allocU = [&](long ne) -> unsigned short* {
    unsigned short* p = (unsigned short*)((char*)d_ws + off);
    off += (((size_t)ne * 2 + 255) / 256) * 256;
    return p;
  };
  unsigned short* DENSE = allocU(V0 * 3);
  float* M0   = allocF(V0);
  float* M1   = allocF(V1);
  float* M2   = allocF(V2);
  float* NC   = allocF(4);     // n = sum(mask) per level
  float* SUMS = allocF(128);   // [0..64) sum, [64..128) sumsq
  float* A0   = allocF(V0 * 16);
  float* U0   = allocF(V0 * 16);
  float* SC0  = allocF(V0 * 16);
  float* CAT0 = allocF(V0 * 32);
  unsigned short* Ta0 = allocU(V0 * 32);
  unsigned short* Tb0 = allocU(V0 * 16);
  float* A1   = allocF(V1 * 32);
  float* U1   = allocF(V1 * 32);
  float* SC1  = allocF(V1 * 32);
  float* CAT1 = allocF(V1 * 64);
  unsigned short* Ta1 = allocU(V1 * 64);
  unsigned short* Tb1 = allocU(V1 * 32);
  float* A2   = allocF(V2 * 48);
  float* U2   = allocF(V2 * 48);
  unsigned short* Ta2 = allocU(V2 * 48);
  unsigned short* Tb2 = allocU(V2 * 48);

  // --- repack all conv weights to bf16 B-fragment layout ---
  struct WDesc { int idx, Co, Ci, K; };
  const WDesc wd[25] = {
    {0, 16, 3, 27},                                            // stem
    {3, 16, 16, 27}, {6, 16, 16, 27}, {9, 16, 16, 27}, {12, 16, 16, 27},
    {15, 32, 16, 16},                                          // down0
    {18, 32, 32, 27}, {21, 32, 32, 27}, {24, 32, 32, 27}, {27, 32, 32, 27},
    {30, 48, 32, 16},                                          // down1
    {33, 48, 48, 27}, {36, 48, 48, 27}, {39, 48, 48, 27}, {42, 48, 48, 27},
    {46, 32, 64, 1}, {49, 32, 64, 27}, {52, 32, 32, 27},       // L1 post b0
    {55, 32, 32, 27}, {58, 32, 32, 27},                        // L1 post b1
    {62, 16, 32, 1}, {65, 16, 32, 27}, {68, 16, 16, 27},       // L0 post b0
    {71, 16, 16, 27}, {74, 16, 16, 27}                         // L0 post b1
  };
  unsigned short* WB[25];
  for (int i = 0; i < 25; ++i) {
    int NCH = (wd[i].Ci + 31) >> 5;
    long nsw = (long)(wd[i].Co / 16) * wd[i].K * NCH * 512;  // swizzled size
    WB[i] = allocU(nsw);
    zero_us<<<gsz(nsw), 256, 0, stream>>>(WB[i], nsw);       // pre-zero padding
    long ne = (long)wd[i].Co * wd[i].Ci * wd[i].K;
    repack_k<<<gsz(ne), 256, 0, stream>>>((const float*)d_in[wd[i].idx], WB[i],
                                          wd[i].Co, wd[i].Ci, wd[i].K, NCH);
  }
  auto P = [&](int i) { return (const float*)d_in[i]; };

  // --- scatter points into dense grid + mask ---
  zero_us<<<gsz(V0 * 3), 256, 0, stream>>>(DENSE, V0 * 3);
  zero_f<<<gsz(V0), 256, 0, stream>>>(M0, V0);
  zero_f<<<1, 32, 0, stream>>>(NC, 4);
  scatter_k<<<gsz(N), 256, 0, stream>>>(F, C, DENSE, M0, N);
  mask_sum_k<<<256, 256, 0, stream>>>(M0, NC + 0, V0);

  // --- stem: 3 -> 16 masked 3x3x3 conv ---
  conv_wmma_t<64, 64, 64, 3, 64, 64, 64, 16, 3, 3, 3, 1, 1, 1, 1, 1, 1>
      <<<dim3(4, 64, 64), 32, 0, stream>>>(DENSE, WB[0], M0, nullptr, A0, 16, 0);

  // --- L0 pre blocks ---
  run_block<64, 64, 64, 16, 16>(A0, Ta0, U0, Tb0, nullptr, nullptr, WB[1], WB[2],
                                P(1), P(2), P(4), P(5), M0, NC + 0, SUMS, A0, stream);
  run_block<64, 64, 64, 16, 16>(A0, Ta0, U0, Tb0, nullptr, nullptr, WB[3], WB[4],
                                P(7), P(8), P(10), P(11), M0, NC + 0, SUMS, A0, stream);

  // --- down 0: bnrelu -> strided conv (4,2,2) -> * m2 ---
  down_mask_k<<<gsz(V1), 256, 0, stream>>>(M0, M1, 64, 64, 16, 32, 32);
  mask_sum_k<<<64, 256, 0, stream>>>(M1, NC + 1, V1);
  stats(A0, SUMS, V0, 16, stream);
  bnApply(A0, Ta0, SUMS, P(13), P(14), M0, NC + 0, V0, 16, stream);
  conv_wmma_t<64, 64, 64, 16, 16, 32, 32, 32, 4, 2, 2, 4, 2, 2, 0, 0, 0>
      <<<dim3(4, 32, 16), 32, 0, stream>>>(Ta0, WB[5], M1, nullptr, A1, 32, 0);

  // --- L1 pre blocks ---
  run_block<16, 32, 32, 32, 32>(A1, Ta1, U1, Tb1, nullptr, nullptr, WB[6], WB[7],
                                P(16), P(17), P(19), P(20), M1, NC + 1, SUMS, A1, stream);
  run_block<16, 32, 32, 32, 32>(A1, Ta1, U1, Tb1, nullptr, nullptr, WB[8], WB[9],
                                P(22), P(23), P(25), P(26), M1, NC + 1, SUMS, A1, stream);

  // --- down 1 ---
  down_mask_k<<<gsz(V2), 256, 0, stream>>>(M1, M2, 32, 32, 4, 16, 16);
  mask_sum_k<<<8, 256, 0, stream>>>(M2, NC + 2, V2);
  stats(A1, SUMS, V1, 32, stream);
  bnApply(A1, Ta1, SUMS, P(28), P(29), M1, NC + 1, V1, 32, stream);
  conv_wmma_t<16, 32, 32, 32, 4, 16, 16, 48, 4, 2, 2, 4, 2, 2, 0, 0, 0>
      <<<dim3(3, 16, 4), 32, 0, stream>>>(Ta1, WB[10], M2, nullptr, A2, 48, 0);

  // --- innermost level blocks (48ch) ---
  run_block<4, 16, 16, 48, 48>(A2, Ta2, U2, Tb2, nullptr, nullptr, WB[11], WB[12],
                               P(31), P(32), P(34), P(35), M2, NC + 2, SUMS, A2, stream);
  run_block<4, 16, 16, 48, 48>(A2, Ta2, U2, Tb2, nullptr, nullptr, WB[13], WB[14],
                               P(37), P(38), P(40), P(41), M2, NC + 2, SUMS, A2, stream);

  // --- up 1: bnrelu -> conv_transpose -> * M1; concat with skip A1 ---
  stats(A2, SUMS, V2, 48, stream);
  bnApply(A2, Ta2, SUMS, P(43), P(44), M2, NC + 2, V2, 48, stream);
  up_conv_k<<<gsz(V1 * 32), 256, 0, stream>>>(Ta2, P(45), M1, CAT1,
                                              16, 16, 48, 16, 32, 32, 32, 64, 32);
  copy_ch_k<<<gsz(V1 * 32), 256, 0, stream>>>(A1, CAT1, V1, 32, 64, 0);

  // --- L1 post blocks (64 -> 32 with nin, then 32 -> 32) ---
  run_block<16, 32, 32, 64, 32>(CAT1, Ta1, U1, Tb1, SC1, WB[15], WB[16], WB[17],
                                P(47), P(48), P(50), P(51), M1, NC + 1, SUMS, A1, stream);
  run_block<16, 32, 32, 32, 32>(A1, Ta1, U1, Tb1, nullptr, nullptr, WB[18], WB[19],
                                P(53), P(54), P(56), P(57), M1, NC + 1, SUMS, A1, stream);

  // --- up 0 ---
  stats(A1, SUMS, V1, 32, stream);
  bnApply(A1, Ta1, SUMS, P(59), P(60), M1, NC + 1, V1, 32, stream);
  up_conv_k<<<gsz(V0 * 16), 256, 0, stream>>>(Ta1, P(61), M0, CAT0,
                                              32, 32, 32, 64, 64, 64, 16, 32, 16);
  copy_ch_k<<<gsz(V0 * 16), 256, 0, stream>>>(A0, CAT0, V0, 16, 32, 0);

  // --- L0 post blocks (32 -> 16 with nin, then 16 -> 16) ---
  run_block<64, 64, 64, 32, 16>(CAT0, Ta0, U0, Tb0, SC0, WB[20], WB[21], WB[22],
                                P(63), P(64), P(66), P(67), M0, NC + 0, SUMS, A0, stream);
  run_block<64, 64, 64, 16, 16>(A0, Ta0, U0, Tb0, nullptr, nullptr, WB[23], WB[24],
                                P(69), P(70), P(72), P(73), M0, NC + 0, SUMS, A0, stream);

  // --- final BN+ReLU (bf16), gather, linear, sigmoid ---
  stats(A0, SUMS, V0, 16, stream);
  bnApply(A0, Tb0, SUMS, P(75), P(76), M0, NC + 0, V0, 16, stream);
  head_k<<<gsz(N), 256, 0, stream>>>(Tb0, C, P(77), P(78), (float*)d_out, N);
}
